// GAEEncoder_36867999269271
// MI455X (gfx1250) — compile-verified, matrix operand order checked by reading, not run on hardware
//
#include <hip/hip_runtime.h>

// ---------------------------------------------------------------------------
// GAE encoder for MI455X (gfx1250): bf16 WMMA GEMMs + L2-resident edge scatter
// GEMM uses direct global b128 fragment loads (A row-major, B pre-transposed)
// so each wmma is fed by pure vector loads -- no LDS staging, no barriers.
// Out-of-range A rows are CLAMPED (not guarded) so the K-loop has no exec
// masking; the store epilogue performs the real bounds check.
// ---------------------------------------------------------------------------

typedef __attribute__((ext_vector_type(16))) __bf16 v16bf;
typedef __attribute__((ext_vector_type(8)))  float  v8f;

union Frag16 { v16bf v; uint4 q[2]; };

// ---------------------------- utility kernels ------------------------------

__global__ void fill_u32_kernel(unsigned* __restrict__ p, unsigned v, size_t n) {
    size_t i  = (size_t)blockIdx.x * blockDim.x + threadIdx.x;
    size_t st = (size_t)gridDim.x * blockDim.x;
    for (; i < n; i += st) p[i] = v;
}

__global__ void cvt_f32_bf16_kernel(const float* __restrict__ in,
                                    __bf16* __restrict__ out, size_t n) {
    size_t i  = (size_t)blockIdx.x * blockDim.x + threadIdx.x;
    size_t st = (size_t)gridDim.x * blockDim.x;
    for (; i < n; i += st) out[i] = (__bf16)in[i];
}

// W[K,N] f32 (row-major) -> WT[N,K] bf16 (row-major over N). Weights are tiny.
__global__ void cvt_t_bf16_kernel(const float* __restrict__ W,
                                  __bf16* __restrict__ WT, int K, int N) {
    size_t total = (size_t)K * N;
    size_t i  = (size_t)blockIdx.x * blockDim.x + threadIdx.x;
    size_t st = (size_t)gridDim.x * blockDim.x;
    for (; i < total; i += st) {
        int k = (int)(i / (size_t)N);
        int n = (int)(i - (size_t)k * N);
        WT[(size_t)n * K + k] = (__bf16)W[i];
    }
}

// ------------------------------ WMMA GEMM ----------------------------------
// C[M,N] = act(A[M,K] @ B[K,N] + bias); A bf16 row-major, BT = B^T bf16 [N,K].
// Fragment layout (ISA 7.12.2, bf16): lane (hg=lane>>4, lr=lane&15) holds two
// contiguous K-groups of 8:  e[0..7] = K hg*8..hg*8+7,  e[8..15] = K 16+hg*8.. .
// => each fragment is two 16B global loads. Each wave computes 32x64 (2x4
// 16x16 tiles), reusing the A fragments across the 4 column tiles.
__global__ __launch_bounds__(256)
void gemm_bf16_wmma_kernel(const __bf16* __restrict__ A,
                           const __bf16* __restrict__ BT,
                           const float*  __restrict__ bias,
                           __bf16*       __restrict__ Cb,
                           int M, int N, int K, int doRelu) {
    const int tid  = threadIdx.x;
    const int lane = tid & 31;
    const int wave = tid >> 5;   // 0..7 : each wave owns a 32-row slab
    const int hg   = lane >> 4;  // K-range selector
    const int lr   = lane & 15;  // row (A) / col (B,C) within a 16x16 tile

    const int rowBase = blockIdx.x * 256 + wave * 32;
    const int colBase = blockIdx.y * 64;

    // Clamped A-row pointers: OOB rows read row M-1 (valid memory); their
    // accumulator rows are discarded by the guarded store epilogue. This keeps
    // the K-loop free of exec-mask branches.
    const __bf16* arow[2];
    #pragma unroll
    for (int rt = 0; rt < 2; ++rt) {
        int row = rowBase + rt * 16 + lr;
        row = row < M ? row : (M - 1);
        arow[rt] = A + (size_t)row * K + hg * 8;
    }
    const __bf16* brow[4];
    #pragma unroll
    for (int ct = 0; ct < 4; ++ct)
        brow[ct] = BT + (size_t)(colBase + ct * 16 + lr) * K + hg * 8;

    v8f acc[2][4];
    #pragma unroll
    for (int rt = 0; rt < 2; ++rt)
        #pragma unroll
        for (int ct = 0; ct < 4; ++ct)
            acc[rt][ct] = (v8f){0.f, 0.f, 0.f, 0.f, 0.f, 0.f, 0.f, 0.f};

    for (int k0 = 0; k0 < K; k0 += 32) {
        Frag16 af[2];
        #pragma unroll
        for (int rt = 0; rt < 2; ++rt) {
            af[rt].q[0] = *(const uint4*)(arow[rt] + k0);
            af[rt].q[1] = *(const uint4*)(arow[rt] + k0 + 16);
        }
        Frag16 bfr[4];
        #pragma unroll
        for (int ct = 0; ct < 4; ++ct) {
            bfr[ct].q[0] = *(const uint4*)(brow[ct] + k0);
            bfr[ct].q[1] = *(const uint4*)(brow[ct] + k0 + 16);
        }
        #pragma unroll
        for (int rt = 0; rt < 2; ++rt)
            #pragma unroll
            for (int ct = 0; ct < 4; ++ct)
                acc[rt][ct] = __builtin_amdgcn_wmma_f32_16x16x32_bf16(
                    false, af[rt].v, false, bfr[ct].v,
                    (short)0, acc[rt][ct], false, false);
    }

    // epilogue: C layout = lane col (lr); VGPR r -> row hg*8+r
    #pragma unroll
    for (int rt = 0; rt < 2; ++rt) {
        #pragma unroll
        for (int ct = 0; ct < 4; ++ct) {
            const int col = colBase + ct * 16 + lr;
            const float bv = bias ? bias[col] : 0.0f;
            #pragma unroll
            for (int r = 0; r < 8; ++r) {
                const int row = rowBase + rt * 16 + hg * 8 + r;
                if (row < M) {
                    float v = acc[rt][ct][r] + bv;
                    if (doRelu) v = fmaxf(v, 0.0f);
                    Cb[(size_t)row * N + col] = (__bf16)v;
                }
            }
        }
    }
}

// ----------------------------- graph kernels -------------------------------

__global__ void deg_accum_kernel(const int* __restrict__ dst,
                                 float* __restrict__ deg, int E) {
    size_t i  = (size_t)blockIdx.x * blockDim.x + threadIdx.x;
    size_t st = (size_t)gridDim.x * blockDim.x;
    for (; i < (size_t)E; i += st) atomicAdd(&deg[dst[i]], 1.0f);
}

__global__ void deg_rsqrt_kernel(const float* __restrict__ deg,
                                 float* __restrict__ dinv, int Nn) {
    size_t i  = (size_t)blockIdx.x * blockDim.x + threadIdx.x;
    size_t st = (size_t)gridDim.x * blockDim.x;
    for (; i < (size_t)Nn; i += st) dinv[i] = rsqrtf(deg[i]);
}

// One wave per (edge + self-loop); lane covers 8 contiguous features (H=256).
// One uint4 (8 bf16) load per lane = full 512B row per wave; f32 atomics into
// agg (both m and agg are L2-resident at these sizes).
__global__ __launch_bounds__(256)
void scatter_edges_kernel(const __bf16* __restrict__ m,
                          const int* __restrict__ src,
                          const int* __restrict__ dst,
                          const float* __restrict__ dinv,
                          float* __restrict__ agg,
                          int E, int Nn, int H) {
    const int lane   = threadIdx.x & 31;
    const int wavesB = blockDim.x >> 5;
    size_t e  = (size_t)blockIdx.x * wavesB + (threadIdx.x >> 5);
    size_t st = (size_t)gridDim.x * wavesB;
    const size_t total = (size_t)E + (size_t)Nn;

    for (; e < total; e += st) {
        int s, d;
        if (e < (size_t)E) { s = src[e]; d = dst[e]; }
        else               { s = d = (int)(e - (size_t)E); }
        const float norm = dinv[s] * dinv[d];

        union { uint4 u; __bf16 b[8]; } pk;
        pk.u = *(const uint4*)(m + (size_t)s * H + lane * 8);
        float* ap = agg + (size_t)d * H + lane * 8;
        #pragma unroll
        for (int j = 0; j < 8; ++j)
            atomicAdd(ap + j, (float)pk.b[j] * norm);
    }
}

// hb = relu(agg + bias) as bf16; 8 elements per thread (2x float4 -> 1x uint4).
__global__ void bias_relu_bf16_kernel(const float* __restrict__ agg,
                                      const float* __restrict__ bias,
                                      __bf16* __restrict__ hb,
                                      int Nn, int H) {
    size_t n8 = ((size_t)Nn * H) >> 3;
    size_t i  = (size_t)blockIdx.x * blockDim.x + threadIdx.x;
    size_t st = (size_t)gridDim.x * blockDim.x;
    const size_t hmask = (size_t)(H - 1);  // H is a power of two (256)
    for (; i < n8; i += st) {
        const size_t base = i << 3;
        const float4 a0 = *(const float4*)(agg + base);
        const float4 a1 = *(const float4*)(agg + base + 4);
        const float* bp = bias + (base & hmask);
        union { uint4 u; __bf16 b[8]; } pk;
        pk.b[0] = (__bf16)fmaxf(a0.x + bp[0], 0.f);
        pk.b[1] = (__bf16)fmaxf(a0.y + bp[1], 0.f);
        pk.b[2] = (__bf16)fmaxf(a0.z + bp[2], 0.f);
        pk.b[3] = (__bf16)fmaxf(a0.w + bp[3], 0.f);
        pk.b[4] = (__bf16)fmaxf(a1.x + bp[4], 0.f);
        pk.b[5] = (__bf16)fmaxf(a1.y + bp[5], 0.f);
        pk.b[6] = (__bf16)fmaxf(a1.z + bp[6], 0.f);
        pk.b[7] = (__bf16)fmaxf(a1.w + bp[7], 0.f);
        *(uint4*)(hb + base) = pk.u;
    }
}

// ------------------------------- pooling -----------------------------------
// thread j = column; blocks stride over rows (coalesced). Values are post-ReLU
// (>= 0), so max can use unsigned atomicMax on the raw float bits.
__global__ __launch_bounds__(256)
void pool_partial_kernel(const __bf16* __restrict__ hb,
                         float* __restrict__ gsum,
                         unsigned* __restrict__ gmax,
                         int Nn, int H) {
    const int j = threadIdx.x;  // H == blockDim.x == 256
    float s = 0.0f, mx = 0.0f;
    for (int r = blockIdx.x; r < Nn; r += gridDim.x) {
        float v = (float)hb[(size_t)r * H + j];
        s += v;
        mx = fmaxf(mx, v);
    }
    atomicAdd(&gsum[j], s);
    atomicMax(&gmax[j], __float_as_uint(mx));
}

// ------------------------------- head MLP ----------------------------------
__global__ __launch_bounds__(256)
void head_mlp_kernel(const float* __restrict__ gsum,
                     const unsigned* __restrict__ gmax,
                     const float* __restrict__ Wp1, const float* __restrict__ bp1,
                     const float* __restrict__ Wp2, const float* __restrict__ bp2,
                     float* __restrict__ out, int Nn, int H, int DE) {
    __shared__ float g[512];
    __shared__ float p[256];
    const int t = threadIdx.x;  // 256 threads, H == 256
    g[t]     = gsum[t] / (float)Nn;
    g[H + t] = __uint_as_float(gmax[t]);
    __syncthreads();

    float a = bp1[t];
    for (int k = 0; k < 2 * H; ++k) a += g[k] * Wp1[(size_t)k * H + t];
    p[t] = fmaxf(a, 0.0f);
    __syncthreads();

    if (t < DE) {
        float o = bp2[t];
        for (int k = 0; k < H; ++k) o += p[k] * Wp2[(size_t)k * DE + t];
        out[t] = o;
    }
}

// ------------------------------- launcher ----------------------------------

extern "C" void kernel_launch(void* const* d_in, const int* in_sizes, int n_in,
                              void* d_out, int out_size, void* d_ws, size_t ws_size,
                              hipStream_t stream) {
    const float* x   = (const float*)d_in[0];
    const int*   ei  = (const int*)  d_in[1];
    const float* W1  = (const float*)d_in[2];
    const float* b1  = (const float*)d_in[3];
    const float* W2  = (const float*)d_in[4];
    const float* b2  = (const float*)d_in[5];
    const float* Wc  = (const float*)d_in[6];
    const float* bc  = (const float*)d_in[7];
    const float* Wp1 = (const float*)d_in[8];
    const float* bp1 = (const float*)d_in[9];
    const float* Wp2 = (const float*)d_in[10];
    const float* bp2 = (const float*)d_in[11];

    const int F  = 64;
    const int H  = in_sizes[3];           // 256
    const int Nn = in_sizes[0] / F;       // 50000
    const int E  = in_sizes[1] / 2;       // 800000
    const int DE = in_sizes[11];          // 128
    const int L  = in_sizes[7] / H;       // 3

    // ---- workspace layout (bf16 activations; tb aliases m: disjoint lifetimes)
    char* ws = (char*)d_ws;
    size_t off = 0;
    auto carve = [&](size_t bytes) -> char* {
        char* p = ws + off;
        off += (bytes + 255) & ~(size_t)255;
        return p;
    };
    __bf16*   xb   = (__bf16*)carve((size_t)Nn * F * 2);
    __bf16*   w1t  = (__bf16*)carve((size_t)H * F * 2);       // W1^T [H,F]
    __bf16*   w2t  = (__bf16*)carve((size_t)H * H * 2);       // W2^T
    __bf16*   wct  = (__bf16*)carve((size_t)L * H * H * 2);   // Wc[l]^T
    __bf16*   hb   = (__bf16*)carve((size_t)Nn * H * 2);
    __bf16*   mb   = (__bf16*)carve((size_t)Nn * H * 2);      // also encoder temp
    float*    agg  = (float*) carve((size_t)Nn * H * 4);
    float*    deg  = (float*) carve((size_t)Nn * 4);
    float*    dinv = (float*) carve((size_t)Nn * 4);
    float*    gsum = (float*) carve((size_t)H * 4);
    unsigned* gmax = (unsigned*)carve((size_t)H * 4);
    __bf16*   tb   = mb;  // encoder intermediate, reuses m buffer

    const int TPB = 256;
    auto gs = [](size_t n, int tpb) { return (unsigned)((n + tpb - 1) / tpb); };

    // ---- 1) convert x + transpose-convert weights to bf16
    cvt_f32_bf16_kernel<<<4096, TPB, 0, stream>>>(x, xb, (size_t)Nn * F);
    cvt_t_bf16_kernel<<<64,  TPB, 0, stream>>>(W1, w1t, F, H);
    cvt_t_bf16_kernel<<<256, TPB, 0, stream>>>(W2, w2t, H, H);
    for (int l = 0; l < L; ++l)
        cvt_t_bf16_kernel<<<256, TPB, 0, stream>>>(
            Wc + (size_t)l * H * H, wct + (size_t)l * H * H, H, H);

    // ---- 2) node encoder: tb = relu(x@W1+b1); hb = tb@W2+b2
    dim3 gg(gs(Nn, 256), H / 64);  // 256 rows/block (8 waves x 32), 64 cols
    gemm_bf16_wmma_kernel<<<gg, TPB, 0, stream>>>(xb, w1t, b1, tb, Nn, H, F, 1);
    gemm_bf16_wmma_kernel<<<gg, TPB, 0, stream>>>(tb, w2t, b2, hb, Nn, H, H, 0);

    // ---- 3) GCN norm: deg = 1 (self loop) + in-degree; dinv = rsqrt(deg)
    fill_u32_kernel<<<gs(Nn, TPB), TPB, 0, stream>>>(
        (unsigned*)deg, 0x3f800000u, (size_t)Nn);
    deg_accum_kernel<<<gs(E, TPB), TPB, 0, stream>>>(ei + E, deg, E);
    deg_rsqrt_kernel<<<gs(Nn, TPB), TPB, 0, stream>>>(deg, dinv, Nn);

    // ---- 4) GCN layers
    const size_t nTotWaves = (size_t)E + Nn;
    const unsigned scatterBlocks = (unsigned)((nTotWaves + 7) / 8);  // 8 waves/blk
    for (int i = 0; i < L; ++i) {
        gemm_bf16_wmma_kernel<<<gg, TPB, 0, stream>>>(
            hb, wct + (size_t)i * H * H, nullptr, mb, Nn, H, H, 0);
        fill_u32_kernel<<<8192, TPB, 0, stream>>>(
            (unsigned*)agg, 0u, (size_t)Nn * H);
        scatter_edges_kernel<<<scatterBlocks, TPB, 0, stream>>>(
            mb, ei, ei + E, dinv, agg, E, Nn, H);
        bias_relu_bf16_kernel<<<4096, TPB, 0, stream>>>(
            agg, bc + (size_t)i * H, hb, Nn, H);
    }

    // ---- 5) pooling + head
    fill_u32_kernel<<<2, TPB, 0, stream>>>((unsigned*)gsum, 0u, (size_t)2 * H);
    pool_partial_kernel<<<512, TPB, 0, stream>>>(hb, gsum, gmax, Nn, H);
    head_mlp_kernel<<<1, TPB, 0, stream>>>(gsum, gmax, Wp1, bp1, Wp2, bp2,
                                           (float*)d_out, Nn, H, DE);
}
